// STGATPredictor_49873160241507
// MI455X (gfx1250) — compile-verified, matrix operand order checked by reading, not run on hardware
//
#include <hip/hip_runtime.h>
#include <math.h>

#define Bc   4
#define Nc   150
#define NEc_ 450
#define Tc   12
#define HIDc 32
#define Hc   4
#define Dc   8
#define Pc   3
#define NEGV -1000000000.0f

typedef __attribute__((ext_vector_type(16))) _Float16 v16h;
typedef __attribute__((ext_vector_type(8)))  float    v8f;

// ---------------- wave32 reductions ----------------
__device__ __forceinline__ float wsum32(float v) {
    #pragma unroll
    for (int o = 16; o >= 1; o >>= 1) v += __shfl_xor(v, o, 32);
    return v;
}
__device__ __forceinline__ float wmax32(float v) {
    #pragma unroll
    for (int o = 16; o >= 1; o >>= 1) v = fmaxf(v, __shfl_xor(v, o, 32));
    return v;
}

// ---------------- WMMA GEMM: C[Mtiles*16,32] = A[M,32] @ W[32,32]^T (+bias) ----------------
// One wave computes a full 16-row x 32-col stripe: A fragment loaded once, two WMMAs.
// Stores are UNGUARDED: C must have Mtiles*16 rows (caller pads buffers).
// A-row reads clamp to Mclamp-1 so no reads past the live region.
// mapmode 0: A row r at r*32 (contiguous)
// mapmode 1: r = bt*Nn + n over tensor laid out (B,Nn,T,32); bt = b*T + t
// mapmode 2: r = b*Nn + n, read slice t = T-1 of (B,Nn,T,32)
__global__ void gemm32_wmma(const float* __restrict__ A, const float* __restrict__ W,
                            const float* __restrict__ bias, float* __restrict__ C,
                            int Mtiles, int Mclamp, int mapmode, int Nn, int T)
{
    int wave = threadIdx.x >> 5;
    int lane = threadIdx.x & 31;
    int tm = blockIdx.x * 4 + wave;
    if (tm >= Mtiles) return;                   // wave-uniform: EXEC stays all-ones
    int half = lane >> 4, lrow = lane & 15;

    int arow = tm * 16 + lrow;
    if (arow >= Mclamp) arow = Mclamp - 1;      // clamp; padded rows never consumed
    long aoff;
    if (mapmode == 0) {
        aoff = (long)arow * 32;
    } else if (mapmode == 1) {
        int bt = arow / Nn, n = arow % Nn;
        int b = bt / T, t = bt % T;
        aoff = (((long)b * Nn + n) * T + t) * 32;
    } else {
        aoff = ((long)arow * T + (T - 1)) * 32;
    }

    // A fragment: 16x32 f16; K split: half0 -> K {0..7,16..23}, half1 -> K {8..15,24..31}
    v16h af, b0, b1;
    #pragma unroll
    for (int j = 0; j < 16; ++j) {
        int k = (j < 8) ? (half * 8 + j) : (8 + half * 8 + j);
        af[j] = (_Float16)A[aoff + k];
        b0[j] = (_Float16)W[lrow * 32 + k];          // output cols 0..15
        b1[j] = (_Float16)W[(16 + lrow) * 32 + k];   // output cols 16..31
    }

    v8f c0 = {}, c1 = {};
    c0 = __builtin_amdgcn_wmma_f32_16x16x32_f16(false, af, false, b0, (short)0, c0, false, false);
    c1 = __builtin_amdgcn_wmma_f32_16x16x32_f16(false, af, false, b1, (short)0, c1, false, false);

    float bv0 = bias ? bias[lrow] : 0.0f;
    float bv1 = bias ? bias[16 + lrow] : 0.0f;
    int rbase = tm * 16 + half * 8;             // C layout: lanes>=16 hold M=r+8
    #pragma unroll
    for (int r = 0; r < 8; ++r) {
        C[(long)(rbase + r) * 32 + lrow]      = c0[r] + bv0;
        C[(long)(rbase + r) * 32 + 16 + lrow] = c1[r] + bv1;
    }
}

// ---------------- GTCN layer: dilated causal conv (gate/filter/skip) + LN + ReLU ----------------
__global__ void gtcn_layer(const float* __restrict__ xin, const float* __restrict__ xres,
                           float* __restrict__ xout,
                           const float* __restrict__ gw, const float* __restrict__ gb,
                           const float* __restrict__ fw, const float* __restrict__ fb,
                           const float* __restrict__ sw, const float* __restrict__ sb,
                           const float* __restrict__ lg, const float* __restrict__ lb,
                           int T, int dil, int P)
{
    __shared__ float xs[4][3][32];
    int wave = threadIdx.x >> 5, lane = threadIdx.x & 31;
    int p = blockIdx.x * 4 + wave;
    bool valid = p < P;
    int pc = valid ? p : 0;
    int t = pc % T, bn = pc / T;

    #pragma unroll
    for (int kk = 0; kk < 3; ++kk) {
        int tt = t - (2 - kk) * dil;            // causal: taps t, t-dil, t-2dil
        xs[wave][kk][lane] = (tt >= 0) ? xin[((long)bn * T + tt) * 32 + lane] : 0.0f;
    }
    __syncthreads();

    int oc = lane;
    float ag = gb[oc], af = fb[oc], as_ = sb[oc];
    for (int ic = 0; ic < 32; ++ic) {
        float x0 = xs[wave][0][ic], x1 = xs[wave][1][ic], x2 = xs[wave][2][ic];
        int wi = (oc * 32 + ic) * 3;
        ag += gw[wi] * x0 + gw[wi + 1] * x1 + gw[wi + 2] * x2;
        af += fw[wi] * x0 + fw[wi + 1] * x1 + fw[wi + 2] * x2;
        as_ += sw[wi] * x0 + sw[wi + 1] * x1 + sw[wi + 2] * x2;
    }
    float g = 1.0f / (1.0f + __expf(-ag));
    float o = g * af + (1.0f - g) * as_;
    float mean = wsum32(o) * (1.0f / 32.0f);
    float d = o - mean;
    float var = wsum32(d * d) * (1.0f / 32.0f);
    float y = d * rsqrtf(var + 1e-5f) * lg[oc] + lb[oc];
    y = fmaxf(y, 0.0f);
    long idx = ((long)bn * T + t) * 32 + oc;
    if (xres) y += xres[idx];
    if (valid) xout[idx] = y;
}

// ---------------- GAT attention scores s_q / s_k ----------------
__global__ void sqk_kernel(const float* __restrict__ q, const float* __restrict__ k,
                           const float* __restrict__ aq, const float* __restrict__ ak,
                           float* __restrict__ sq, float* __restrict__ sk, int total)
{
    int i = blockIdx.x * blockDim.x + threadIdx.x;
    if (i >= total * 4) return;
    int h = i & 3, row = i >> 2;
    float s1 = 0.0f, s2 = 0.0f;
    #pragma unroll
    for (int d = 0; d < 8; ++d) {
        s1 += q[(long)row * 32 + h * 8 + d] * aq[h * 8 + d];
        s2 += k[(long)row * 32 + h * 8 + d] * ak[h * 8 + d];
    }
    sq[i] = s1; sk[i] = s2;
}

// ---------------- edge score vals[bt,e,h] = len*w0[h] + speed*w1[h] ----------------
__global__ void vals_kernel(const float* __restrict__ We, const float* __restrict__ ae,
                            const float* __restrict__ elen, const float* __restrict__ speed,
                            float* __restrict__ vals, int total, int NE, int T)
{
    int i = blockIdx.x * blockDim.x + threadIdx.x;
    if (i >= total) return;
    int h = i & 3, rest = i >> 2;
    int e = rest % NE, bt = rest / NE;
    int b = bt / T, t = bt % T;
    float w0 = 0.0f, w1 = 0.0f;
    #pragma unroll
    for (int d = 0; d < 8; ++d) {
        float a = ae[h * 8 + d];
        w0 += We[(h * 8 + d) * 2 + 0] * a;
        w1 += We[(h * 8 + d) * 2 + 1] * a;
    }
    float sp = speed[((long)b * NE + e) * T + t];
    vals[i] = elen[e] * w0 + sp * w1;
}

// ---------------- fused GAT attention: softmax over j + alpha@V + ELU ----------------
// mode 0: no mask (adaptive adj, softmax output strictly > 0)
// mode 1: fixed adj (eye | edge), s_e injected from eid/vals
// mode 2: line graph mask dst[i]==src[j]
__global__ void attn_kernel(const float* __restrict__ sq, const float* __restrict__ sk,
                            const float* __restrict__ v, float* __restrict__ xout,
                            const int* __restrict__ eid, const float* __restrict__ vals,
                            const int* __restrict__ ei,
                            int Nn, int T, int mode, int NE)
{
    int wave = threadIdx.x >> 5, lane = threadIdx.x & 31;
    int blk = blockIdx.x;
    int i = blk % Nn, bt = blk / Nn;
    int h = wave;
    float sqi = sq[((long)bt * Nn + i) * 4 + h];
    int dst_i = (mode == 2) ? ei[2 * i + 1] : 0;

    float m = -3.0e38f;
    for (int j = lane; j < Nn; j += 32) {
        float s = sqi + sk[((long)bt * Nn + j) * 4 + h];
        bool ok = true;
        if (mode == 1) {
            int e = eid[i * Nn + j];
            if (e) s += vals[((long)bt * NE + (e - 1)) * 4 + h];
            ok = (i == j) || (e != 0);
        } else if (mode == 2) {
            ok = (dst_i == ei[2 * j]);
        }
        s = (s >= 0.0f) ? s : 0.2f * s;
        if (!ok) s = NEGV;
        m = fmaxf(m, s);
    }
    m = wmax32(m);

    float l = 0.0f;
    float acc[8];
    #pragma unroll
    for (int d = 0; d < 8; ++d) acc[d] = 0.0f;
    for (int j = lane; j < Nn; j += 32) {
        float s = sqi + sk[((long)bt * Nn + j) * 4 + h];
        bool ok = true;
        if (mode == 1) {
            int e = eid[i * Nn + j];
            if (e) s += vals[((long)bt * NE + (e - 1)) * 4 + h];
            ok = (i == j) || (e != 0);
        } else if (mode == 2) {
            ok = (dst_i == ei[2 * j]);
        }
        s = (s >= 0.0f) ? s : 0.2f * s;
        if (!ok) s = NEGV;
        float ex = __expf(s - m);
        l += ex;
        const float* vj = v + ((long)bt * Nn + j) * 32 + h * 8;
        #pragma unroll
        for (int d = 0; d < 8; ++d) acc[d] += ex * vj[d];
    }
    l = wsum32(l);
    #pragma unroll
    for (int d = 0; d < 8; ++d) acc[d] = wsum32(acc[d]);

    if (lane < 8) {
        float o = acc[lane] / l;
        o = (o > 0.0f) ? o : (__expf(o) - 1.0f);      // ELU
        int b = bt / T, t = bt % T;
        xout[(((long)b * Nn + i) * T + t) * 32 + h * 8 + lane] = o;
    }
}

// ---------------- small elementwise kernels ----------------
__global__ void nodeproj_kernel(const float* __restrict__ ns, const float* __restrict__ w,
                                const float* __restrict__ b, float* __restrict__ out, int rows)
{
    long i = (long)blockIdx.x * blockDim.x + threadIdx.x;
    if (i >= (long)rows * 32) return;
    int c = (int)(i & 31); long r = i >> 5;
    out[i] = ns[r * 2] * w[c * 2] + ns[r * 2 + 1] * w[c * 2 + 1] + b[c];
}
__global__ void edgeproj_kernel(const float* __restrict__ sp, const float* __restrict__ w,
                                const float* __restrict__ b, float* __restrict__ out, long rows)
{
    long i = (long)blockIdx.x * blockDim.x + threadIdx.x;
    if (i >= rows * 32) return;
    int c = (int)(i & 31); long r = i >> 5;
    out[i] = sp[r] * w[c] + b[c];
}
__global__ void eid_clear(int* __restrict__ eid, int n)
{
    int i = blockIdx.x * blockDim.x + threadIdx.x;
    if (i < n) eid[i] = 0;
}
__global__ void eid_set(const int* __restrict__ ei, int* __restrict__ eid, int ne, int n)
{
    int e = blockIdx.x * blockDim.x + threadIdx.x;
    if (e < ne) eid[ei[2 * e] * n + ei[2 * e + 1]] = e + 1;
}
__global__ void hsum_kernel(const float* __restrict__ xf, const float* __restrict__ xa,
                            float* __restrict__ hs, int total, int T)
{
    int i = blockIdx.x * blockDim.x + threadIdx.x;
    if (i >= total) return;
    int c = i & 31, r = i >> 5;
    long off = ((long)r * T + (T - 1)) * 32 + c;
    hs[i] = xf[off] + xa[off];
}
__global__ void fuse_kernel(const float* __restrict__ g1, const float* __restrict__ u2,
                            const float* __restrict__ u3, float* __restrict__ hn, int total)
{
    int i = blockIdx.x * blockDim.x + threadIdx.x;
    if (i >= total) return;
    float g = 1.0f / (1.0f + __expf(-g1[i]));
    hn[i] = tanhf(g * u2[i] + (1.0f - g) * u3[i]);
}
__global__ void head_kernel(const float* __restrict__ h, const float* __restrict__ w,
                            const float* __restrict__ b, float* __restrict__ out,
                            int rows, int mode, int T)
{
    int i = blockIdx.x * blockDim.x + threadIdx.x;
    if (i >= rows * 3) return;
    int p = i % 3, r = i / 3;
    const float* hr = (mode == 0) ? h + (long)r * 32 : h + ((long)r * T + (T - 1)) * 32;
    float s = b[p];
    #pragma unroll
    for (int c = 0; c < 32; ++c) s += hr[c] * w[p * 32 + c];
    out[i] = s;
}

// ---------------- host-side parameter parsing (sorted pytree leaf order) ----------------
struct GtcnL { const float *fb, *fw, *gb, *gw, *lb, *lg, *sb, *sw; };
struct GatP  { const float *We, *Wk, *Wq, *Wv, *ae, *ak, *aq; };
struct BlockP { GatP g; GtcnL l[2]; };

static void parse_block(void* const* din, int& idx, BlockP& b, bool edge)
{
    auto F = [&](int i) { return (const float*)din[i]; };
    b.g.We = nullptr; b.g.ae = nullptr;
    if (edge) b.g.We = F(idx++);
    b.g.Wk = F(idx++); b.g.Wq = F(idx++); b.g.Wv = F(idx++);
    if (edge) b.g.ae = F(idx++);
    b.g.ak = F(idx++); b.g.aq = F(idx++);
    for (int L = 0; L < 2; ++L) {
        GtcnL& y = b.l[L];
        y.fb = F(idx++); y.fw = F(idx++); y.gb = F(idx++); y.gw = F(idx++);
        y.lb = F(idx++); y.lg = F(idx++); y.sb = F(idx++); y.sw = F(idx++);
    }
}

extern "C" void kernel_launch(void* const* d_in, const int* in_sizes, int n_in,
                              void* d_out, int out_size, void* d_ws, size_t ws_size,
                              hipStream_t stream)
{
    (void)in_sizes; (void)n_in; (void)out_size; (void)ws_size;
    auto F = [&](int i) { return (const float*)d_in[i]; };

    const float* node_seq  = F(0);
    const float* speed_seq = F(1);
    const int*   edge_idx  = (const int*)d_in[2];
    const float* edge_len  = F(3);

    int idx = 4;
    BlockP adp[2]; parse_block(d_in, idx, adp[0], false); parse_block(d_in, idx, adp[1], false);
    const float* db_ = F(idx++);
    const float* dw_ = F(idx++);
    BlockP edg[2]; parse_block(d_in, idx, edg[0], false); parse_block(d_in, idx, edg[1], false);
    const float* epb = F(idx++);
    const float* epw = F(idx++);
    idx++; /* emb_dst: dead (softmax adj always > 0) */
    idx++; /* emb_src: dead */
    BlockP fix[2]; parse_block(d_in, idx, fix[0], true); parse_block(d_in, idx, fix[1], true);
    const float* W1 = F(idx++); const float* W2 = F(idx++); const float* W3 = F(idx++);
    const float* b1 = F(idx++); const float* b2 = F(idx++); const float* b3 = F(idx++);
    const float* npb = F(idx++);
    const float* npw = F(idx++);
    const float* sbp = F(idx++);
    const float* swp = F(idx++);
    const float* vbp = F(idx++);
    const float* vwp = F(idx++);

    // ---------------- workspace layout (all resident in L2; ~17 MB) ----------------
    const long BIGF = (long)Bc * NEc_ * Tc * HIDc;        // 691200 floats
    const long NODF = (long)Bc * Nc * Tc * HIDc;          // 230400 floats
    const long SQF  = (long)Bc * Tc * NEc_ * Hc;          // 86400 floats
    const long FUSF = 640L * 32;                          // fusion rows padded to tile multiple
    float* ws = (float*)d_ws;
    float* tA = ws;                 // GTCN out / GEMM A
    float* tB = tA + BIGF;          // GTCN tmp / Q
    float* tC = tB + BIGF;          // K
    float* tD = tC + BIGF;          // V
    float* x_fix = tD + BIGF;
    float* x_adp = x_fix + NODF;
    float* x_edg = x_adp + NODF;
    float* sqb   = x_edg + BIGF;
    float* skb   = sqb + SQF;
    float* valsb = skb + SQF;
    float* hsb   = valsb + SQF;
    float* g1b   = hsb + FUSF;
    float* u2b   = g1b + FUSF;
    float* u3b   = u2b + FUSF;
    float* hnb   = u3b + FUSF;
    int*   eidb  = (int*)(hnb + FUSF);

    // GEMM launch helper: C gets Mtiles*16 rows (padded), A reads clamp to Mclamp rows
    auto gemm = [&](const float* A, const float* Wm, const float* bias, float* Cc,
                    int M, int mapmode, int Nn) {
        int Mt = (M + 15) / 16;
        gemm32_wmma<<<(Mt + 3) / 4, 128, 0, stream>>>(A, Wm, bias, Cc, Mt, M, mapmode, Nn, Tc);
    };

    // ---------------- setup ----------------
    eid_clear<<<(Nc * Nc + 255) / 256, 256, 0, stream>>>(eidb, Nc * Nc);
    eid_set<<<(NEc_ + 255) / 256, 256, 0, stream>>>(edge_idx, eidb, NEc_, Nc);
    {
        long rows = (long)Bc * Nc * Tc;
        int g = (int)((rows * 32 + 255) / 256);
        nodeproj_kernel<<<g, 256, 0, stream>>>(node_seq, npw, npb, x_fix, (int)rows);
        nodeproj_kernel<<<g, 256, 0, stream>>>(node_seq, npw, npb, x_adp, (int)rows);
    }
    {
        long rows = (long)Bc * NEc_ * Tc;
        int g = (int)((rows * 32 + 255) / 256);
        edgeproj_kernel<<<g, 256, 0, stream>>>(speed_seq, epw, epb, x_edg, rows);
    }

    // ---------------- path runner ----------------
    auto run_path = [&](float* x, int Nn, BlockP* blks, int mode) {
        long P = (long)Bc * Nn * Tc;
        int  M = (int)((long)Bc * Tc * Nn);       // multiple of 16 for both N values
        int  gridT = (int)((P + 3) / 4);
        for (int bi = 0; bi < 2; ++bi) {
            BlockP& bl = blks[bi];
            // GTCN: layer0 (dil=1) x->tB ; layer1 (dil=2) tB->tA with residual x
            gtcn_layer<<<gridT, 128, 0, stream>>>(x, nullptr, tB,
                bl.l[0].gw, bl.l[0].gb, bl.l[0].fw, bl.l[0].fb,
                bl.l[0].sw, bl.l[0].sb, bl.l[0].lg, bl.l[0].lb, Tc, 1, (int)P);
            gtcn_layer<<<gridT, 128, 0, stream>>>(tB, x, tA,
                bl.l[1].gw, bl.l[1].gb, bl.l[1].fw, bl.l[1].fb,
                bl.l[1].sw, bl.l[1].sb, bl.l[1].lg, bl.l[1].lb, Tc, 2, (int)P);
            // Q/K/V projections via WMMA (gather from (B,Nn,T,32) layout)
            gemm(tA, bl.g.Wq, nullptr, tB, M, 1, Nn);
            gemm(tA, bl.g.Wk, nullptr, tC, M, 1, Nn);
            gemm(tA, bl.g.Wv, nullptr, tD, M, 1, Nn);
            sqk_kernel<<<(M * 4 + 255) / 256, 256, 0, stream>>>(tB, tC, bl.g.aq, bl.g.ak,
                                                                sqb, skb, M);
            if (mode == 1) {
                int tot = Bc * Tc * NEc_ * Hc;
                vals_kernel<<<(tot + 255) / 256, 256, 0, stream>>>(bl.g.We, bl.g.ae,
                    edge_len, speed_seq, valsb, tot, NEc_, Tc);
            }
            attn_kernel<<<M, 128, 0, stream>>>(sqb, skb, tD, x, eidb, valsb, edge_idx,
                                               Nn, Tc, mode, NEc_);
        }
    };

    run_path(x_fix, Nc, fix, 1);     // fixed adjacency + edge features
    run_path(x_adp, Nc, adp, 0);     // adaptive adjacency: mask is a no-op
    run_path(x_edg, NEc_, edg, 2);   // line graph

    // ---------------- fusion + heads ----------------
    hsum_kernel<<<(600 * 32 + 255) / 256, 256, 0, stream>>>(x_fix, x_adp, hsb, 600 * 32, Tc);
    gemm(hsb,   W1, b1, g1b, 600, 0, Nc);
    gemm(x_fix, W2, b2, u2b, 600, 2, Nc);
    gemm(x_adp, W3, b3, u3b, 600, 2, Nc);
    fuse_kernel<<<(600 * 32 + 255) / 256, 256, 0, stream>>>(g1b, u2b, u3b, hnb, 600 * 32);

    float* out = (float*)d_out;
    head_kernel<<<(600 * 3 + 255) / 256, 256, 0, stream>>>(hnb, dw_, db_, out, 600, 0, Tc);
    head_kernel<<<(600 * 3 + 255) / 256, 256, 0, stream>>>(hnb, swp, sbp, out + 1800, 600, 0, Tc);
    head_kernel<<<(1800 * 3 + 255) / 256, 256, 0, stream>>>(x_edg, vwp, vbp, out + 3600, 1800, 1, Tc);
}
